// OursLayer3_51213190038000
// MI455X (gfx1250) — compile-verified, matrix operand order checked by reading, not run on hardware
//
#include <hip/hip_runtime.h>

// Problem sizes (from reference)
#define N_S   8192   // Sinput rows
#define M_R   4096   // Rinput rows
#define FIN   512
#define FOUT  256
#define ALPHA 0.2f
#define BN_EPS 1e-5f
#define NEG_BIG (-9.0e15f)

typedef __attribute__((ext_vector_type(16))) __bf16 v16bf;
typedef __attribute__((ext_vector_type(8)))  __bf16 v8bf;
typedef __attribute__((ext_vector_type(8)))  float  v8f;
typedef __attribute__((ext_vector_type(4)))  float  v4f;
typedef __attribute__((ext_vector_type(4)))  int    v4i;

__device__ __forceinline__ __bf16 f2bf(float f) {
  unsigned u; __builtin_memcpy(&u, &f, 4);
  u += 0x7fffu + ((u >> 16) & 1u);              // round-to-nearest-even
  unsigned short h = (unsigned short)(u >> 16);
  __bf16 r; __builtin_memcpy(&r, &h, 2);
  return r;
}

__device__ __forceinline__ float leaky(float x) { return x > 0.f ? x : ALPHA * x; }

// Concatenate two contiguous 8-element runs into a WMMA A/B register frag.
__device__ __forceinline__ v16bf cat16(v8bf lo, v8bf hi) {
  return __builtin_shufflevector(lo, hi, 0, 1, 2, 3, 4, 5, 6, 7,
                                         8, 9, 10, 11, 12, 13, 14, 15);
}
// Load a full frag from a contiguous 24-element window: runs [0..7] and [16..23].
__device__ __forceinline__ v16bf load_frag(const __bf16* p) {
  return cat16(*(const v8bf*)p, *(const v8bf*)(p + 16));
}

// ---------------------------------------------------------------- f32 -> bf16
__global__ void k_cvt_bf16(const float* __restrict__ x, __bf16* __restrict__ y, int n) {
  for (int i = blockIdx.x * blockDim.x + threadIdx.x; i < n; i += gridDim.x * blockDim.x)
    y[i] = f2bf(x[i]);
}

// ------------------------- W (FIN x FOUT) -> Wt (FOUT x FIN) bf16 transpose
__global__ void k_cvt_wT(const float* __restrict__ W, __bf16* __restrict__ Wt) {
  for (int idx = blockIdx.x * blockDim.x + threadIdx.x; idx < FIN * FOUT;
       idx += gridDim.x * blockDim.x) {
    const int k = idx >> 8, n = idx & (FOUT - 1);
    Wt[(size_t)n * FIN + k] = f2bf(W[idx]);
  }
}

// ------------------------------------------ H = X(rows x 512) @ W(512 x 256)
// Wt is column-major (FOUT x FIN) so B-frags are contiguous b128 loads.
// wave: 4 M-tiles x 1 N-tile. Writes H f32 row-major + H bf16 TRANSPOSED.
__global__ void k_gemm_xw(const __bf16* __restrict__ A, const __bf16* __restrict__ Wt,
                          float* __restrict__ C, __bf16* __restrict__ CbT, int rows) {
  const int wave = threadIdx.x >> 5, lane = threadIdx.x & 31;
  const int r = lane & 15;
  const int koff = (lane & 16) ? 8 : 0;
  const int hioff = (lane & 16) ? 8 : 0;
  const int mt = blockIdx.x * 64;
  const int nt = (blockIdx.y * 4 + wave) * 16;
  v8f acc[4] = {};
  for (int k0 = 0; k0 < FIN; k0 += 32) {
    const __bf16* Bcol = Wt + (size_t)(nt + r) * FIN + k0 + koff;
    const v16bf b = load_frag(Bcol);
#pragma unroll
    for (int t = 0; t < 4; ++t) {
      const __bf16* Arow = A + (size_t)(mt + t * 16 + r) * FIN + k0 + koff;
      __builtin_prefetch(Arow + 32, 0, 0);
      acc[t] = __builtin_amdgcn_wmma_f32_16x16x32_bf16(false, load_frag(Arow), false, b,
                                                       (short)0, acc[t], false, false);
    }
  }
#pragma unroll
  for (int t = 0; t < 4; ++t) {
    v8bf pk;
#pragma unroll
    for (int v = 0; v < 8; ++v) {
      const int row = mt + t * 16 + hioff + v;
      C[(size_t)row * FOUT + nt + r] = acc[t][v];
      pk[v] = f2bf(acc[t][v]);
    }
    // transposed bf16 copy: 8 consecutive rows per lane -> one 16B store
    *(v8bf*)&CbT[(size_t)(nt + r) * rows + mt + t * 16 + hioff] = pk;
  }
}

// ----------------------------------------------------- s[i] = H[i,:] . avec
__global__ void k_rowdot(const float* __restrict__ H, const float* __restrict__ avec,
                         float* __restrict__ s, int rows) {
  const int wave = threadIdx.x >> 5, lane = threadIdx.x & 31;
  const int row = blockIdx.x * (blockDim.x >> 5) + wave;
  if (row >= rows) return;
  const int f0 = lane * 8;
  const v4f h0 = *(const v4f*)&H[(size_t)row * FOUT + f0];
  const v4f h1 = *(const v4f*)&H[(size_t)row * FOUT + f0 + 4];
  const v4f a0 = *(const v4f*)&avec[f0];
  const v4f a1 = *(const v4f*)&avec[f0 + 4];
  float acc = h0[0]*a0[0] + h0[1]*a0[1] + h0[2]*a0[2] + h0[3]*a0[3]
            + h1[0]*a1[0] + h1[1]*a1[1] + h1[2]*a1[2] + h1[3]*a1[3];
#pragma unroll
  for (int off = 16; off; off >>= 1) acc += __shfl_xor(acc, off, 32);
  if (lane == 0) s[row] = acc;
}

// -------------------- per-row softmax stats over e(i,j) (masked leaky logits)
__global__ void k_softmax_stats(const int* __restrict__ adj, const float* __restrict__ s1,
                                const float* __restrict__ s2, float* __restrict__ rm,
                                float* __restrict__ rl) {
  __shared__ float red[256];
  const int i = blockIdx.x, tid = threadIdx.x;
  const float s2i = s2[i];
  const int j0 = tid * 16;                         // contiguous 16-int run
  float e[16];
  float mloc = -3.0e38f;
#pragma unroll
  for (int c = 0; c < 4; ++c) {
    const v4i ad = *(const v4i*)&adj[(size_t)i * M_R + j0 + 4 * c];
    const v4f sv = *(const v4f*)&s1[j0 + 4 * c];
#pragma unroll
    for (int u = 0; u < 4; ++u) {
      const float v = (ad[u] > 0) ? leaky(s2i + sv[u]) : NEG_BIG;
      e[4 * c + u] = v;
      mloc = fmaxf(mloc, v);
    }
  }
  red[tid] = mloc; __syncthreads();
  for (int o = 128; o; o >>= 1) { if (tid < o) red[tid] = fmaxf(red[tid], red[tid + o]); __syncthreads(); }
  const float mrow = red[0]; __syncthreads();
  float sl = 0.f;
#pragma unroll
  for (int t = 0; t < 16; ++t) sl += __expf(e[t] - mrow);
  red[tid] = sl; __syncthreads();
  for (int o = 128; o; o >>= 1) { if (tid < o) red[tid] += red[tid + o]; __syncthreads(); }
  if (tid == 0) { rm[i] = mrow; rl[i] = red[0]; }
}

// ------------------- u_pre = att @ h1; wave: 1 M-tile x 4 N-tiles, att once
// h1bT is (FOUT x M_R): B-frags are contiguous b128 loads.
__global__ void k_att_h1(const int* __restrict__ adj, const float* __restrict__ s1,
                         const float* __restrict__ s2, const float* __restrict__ rm,
                         const float* __restrict__ rl, const __bf16* __restrict__ h1bT,
                         float* __restrict__ up) {
  const int wave = threadIdx.x >> 5, lane = threadIdx.x & 31;
  const int r = lane & 15;
  const int koff = (lane & 16) ? 8 : 0;
  const int ibase = blockIdx.x * 16;
  const int i = ibase + r;
  const float s2i = s2[i], mi = rm[i], linv = 1.f / rl[i];
  v8f acc[4] = {};
  for (int j0 = 0; j0 < M_R; j0 += 32) {
    const int jlo = j0 + koff;
    const v4i ad0 = *(const v4i*)&adj[(size_t)i * M_R + jlo];
    const v4i ad1 = *(const v4i*)&adj[(size_t)i * M_R + jlo + 4];
    const v4i ad2 = *(const v4i*)&adj[(size_t)i * M_R + jlo + 16];
    const v4i ad3 = *(const v4i*)&adj[(size_t)i * M_R + jlo + 20];
    const v4f sv0 = *(const v4f*)&s1[jlo];
    const v4f sv1 = *(const v4f*)&s1[jlo + 4];
    const v4f sv2 = *(const v4f*)&s1[jlo + 16];
    const v4f sv3 = *(const v4f*)&s1[jlo + 20];
    v16bf a;
#pragma unroll
    for (int u = 0; u < 4; ++u) {
      float e0 = (ad0[u] > 0) ? leaky(s2i + sv0[u]) : NEG_BIG;
      float e1 = (ad1[u] > 0) ? leaky(s2i + sv1[u]) : NEG_BIG;
      float e2 = (ad2[u] > 0) ? leaky(s2i + sv2[u]) : NEG_BIG;
      float e3 = (ad3[u] > 0) ? leaky(s2i + sv3[u]) : NEG_BIG;
      a[u]      = f2bf(__expf(e0 - mi) * linv);
      a[4 + u]  = f2bf(__expf(e1 - mi) * linv);
      a[8 + u]  = f2bf(__expf(e2 - mi) * linv);
      a[12 + u] = f2bf(__expf(e3 - mi) * linv);
    }
#pragma unroll
    for (int t = 0; t < 4; ++t) {
      const int n = (wave * 4 + t) * 16 + r;
      const v16bf b = load_frag(h1bT + (size_t)n * M_R + jlo);
      acc[t] = __builtin_amdgcn_wmma_f32_16x16x32_bf16(false, a, false, b, (short)0,
                                                       acc[t], false, false);
    }
  }
#pragma unroll
  for (int t = 0; t < 4; ++t)
#pragma unroll
    for (int v = 0; v < 8; ++v)
      up[(size_t)(ibase + v + ((lane & 16) ? 8 : 0)) * FOUT + (wave * 4 + t) * 16 + r] =
          acc[t][v];
}

// ------------------- v_pre = att.T @ h2; wave: 1 M-tile x 4 N-tiles
// h2bT is (FOUT x N_S): B-frags are contiguous b128 loads.
__global__ void k_attT_h2(const int* __restrict__ adj, const float* __restrict__ s1,
                          const float* __restrict__ s2, const float* __restrict__ rm,
                          const float* __restrict__ rl, const __bf16* __restrict__ h2bT,
                          float* __restrict__ vp) {
  const int wave = threadIdx.x >> 5, lane = threadIdx.x & 31;
  const int r = lane & 15;
  const int koff = (lane & 16) ? 8 : 0;
  const int jbase = blockIdx.x * 16;
  const int j = jbase + r;
  const float s1j = s1[j];
  v8f acc[4] = {};
  for (int i0 = 0; i0 < N_S; i0 += 32) {
    const int ilo = i0 + koff;
    const v4f s20 = *(const v4f*)&s2[ilo];
    const v4f s21 = *(const v4f*)&s2[ilo + 4];
    const v4f s22 = *(const v4f*)&s2[ilo + 16];
    const v4f s23 = *(const v4f*)&s2[ilo + 20];
    const v4f rm0 = *(const v4f*)&rm[ilo];
    const v4f rm1 = *(const v4f*)&rm[ilo + 4];
    const v4f rm2 = *(const v4f*)&rm[ilo + 16];
    const v4f rm3 = *(const v4f*)&rm[ilo + 20];
    const v4f rl0 = *(const v4f*)&rl[ilo];
    const v4f rl1 = *(const v4f*)&rl[ilo + 4];
    const v4f rl2 = *(const v4f*)&rl[ilo + 16];
    const v4f rl3 = *(const v4f*)&rl[ilo + 20];
    v16bf a;
#pragma unroll
    for (int u = 0; u < 4; ++u) {
      const int iA = ilo + u, iB = ilo + 4 + u, iC = ilo + 16 + u, iD = ilo + 20 + u;
      float e0 = (adj[(size_t)iA * M_R + j] > 0) ? leaky(s20[u] + s1j) : NEG_BIG;
      float e1 = (adj[(size_t)iB * M_R + j] > 0) ? leaky(s21[u] + s1j) : NEG_BIG;
      float e2 = (adj[(size_t)iC * M_R + j] > 0) ? leaky(s22[u] + s1j) : NEG_BIG;
      float e3 = (adj[(size_t)iD * M_R + j] > 0) ? leaky(s23[u] + s1j) : NEG_BIG;
      a[u]      = f2bf(__expf(e0 - rm0[u]) / rl0[u]);
      a[4 + u]  = f2bf(__expf(e1 - rm1[u]) / rl1[u]);
      a[8 + u]  = f2bf(__expf(e2 - rm2[u]) / rl2[u]);
      a[12 + u] = f2bf(__expf(e3 - rm3[u]) / rl3[u]);
    }
#pragma unroll
    for (int t = 0; t < 4; ++t) {
      const int n = (wave * 4 + t) * 16 + r;
      const v16bf b = load_frag(h2bT + (size_t)n * N_S + ilo);
      acc[t] = __builtin_amdgcn_wmma_f32_16x16x32_bf16(false, a, false, b, (short)0,
                                                       acc[t], false, false);
    }
  }
#pragma unroll
  for (int t = 0; t < 4; ++t)
#pragma unroll
    for (int v = 0; v < 8; ++v)
      vp[(size_t)(jbase + v + ((lane & 16) ? 8 : 0)) * FOUT + (wave * 4 + t) * 16 + r] =
          acc[t][v];
}

// --------------------------------------------- BN column stats (mean, biased var)
__global__ void k_bn_stats(const float* __restrict__ X, int rows,
                           float* __restrict__ mean, float* __restrict__ var) {
  __shared__ float rs[256], rq[256];
  const int f = blockIdx.x, tid = threadIdx.x;
  float s = 0.f, q = 0.f;
  for (int rI = tid; rI < rows; rI += blockDim.x) {
    const float x = X[(size_t)rI * FOUT + f];
    s += x; q += x * x;
  }
  rs[tid] = s; rq[tid] = q; __syncthreads();
  for (int o = 128; o; o >>= 1) {
    if (tid < o) { rs[tid] += rs[tid + o]; rq[tid] += rq[tid + o]; }
    __syncthreads();
  }
  if (tid == 0) {
    const float mu = rs[0] / (float)rows;
    mean[f] = mu;
    var[f]  = rq[0] / (float)rows - mu * mu;
  }
}

// ---------------------------------------- y = leaky(bn(x)) as bf16
__global__ void k_bn_apply(const float* __restrict__ X, int n,
                           const float* __restrict__ mean, const float* __restrict__ var,
                           const float* __restrict__ gamma, const float* __restrict__ beta,
                           __bf16* __restrict__ Y) {
  for (int idx = blockIdx.x * blockDim.x + threadIdx.x; idx < n;
       idx += gridDim.x * blockDim.x) {
    const int f = idx & (FOUT - 1);
    const float y = (X[idx] - mean[f]) * rsqrtf(var[f] + BN_EPS) * gamma[f] + beta[f];
    Y[idx] = f2bf(leaky(y));
  }
}

// ------------------------------ out = elu(u_out @ v_out.T), (8192 x 4096) f32
// wave: 1 M-tile x 4 N-tiles; A and B frags are contiguous 16B runs
__global__ void k_uvT_elu(const __bf16* __restrict__ U, const __bf16* __restrict__ V,
                          float* __restrict__ out) {
  const int wave = threadIdx.x >> 5, lane = threadIdx.x & 31;
  const int r = lane & 15;
  const int koff = (lane & 16) ? 8 : 0;
  const int ibase = blockIdx.x * 16;
  v8f acc[4] = {};
#pragma unroll 2
  for (int k0 = 0; k0 < FOUT; k0 += 32) {
    const v16bf a = load_frag(U + (size_t)(ibase + r) * FOUT + k0 + koff);
#pragma unroll
    for (int t = 0; t < 4; ++t) {
      const int nb = (blockIdx.y * 16 + wave * 4 + t) * 16;
      const v16bf b = load_frag(V + (size_t)(nb + r) * FOUT + k0 + koff);
      acc[t] = __builtin_amdgcn_wmma_f32_16x16x32_bf16(false, a, false, b, (short)0,
                                                       acc[t], false, false);
    }
  }
#pragma unroll
  for (int t = 0; t < 4; ++t) {
    const int nb = (blockIdx.y * 16 + wave * 4 + t) * 16;
#pragma unroll
    for (int v = 0; v < 8; ++v) {
      const float x = acc[t][v];
      out[(size_t)(ibase + v + ((lane & 16) ? 8 : 0)) * M_R + nb + r] =
          x > 0.f ? x : __expf(x) - 1.f;
    }
  }
}

extern "C" void kernel_launch(void* const* d_in, const int* in_sizes, int n_in,
                              void* d_out, int out_size, void* d_ws, size_t ws_size,
                              hipStream_t stream) {
  const float* Sinput = (const float*)d_in[0];
  const float* Rinput = (const float*)d_in[1];
  const int*   adj    = (const int*)d_in[2];
  const float* W1     = (const float*)d_in[6];
  const float* W2     = (const float*)d_in[7];
  const float* avec   = (const float*)d_in[8];   // (512,1): a1 = [0:256), a2 = [256:512)
  const float* bn1_g  = (const float*)d_in[9];
  const float* bn1_b  = (const float*)d_in[10];
  const float* bn2_g  = (const float*)d_in[11];
  const float* bn2_b  = (const float*)d_in[12];
  float* out = (float*)d_out;

  char* ws = (char*)d_ws;
  size_t o = 0;
  auto take = [&](size_t bytes) {
    char* p = ws + o;
    o += (bytes + 255) & ~(size_t)255;
    return p;
  };
  __bf16* Sb    = (__bf16*)take((size_t)N_S * FIN * 2);
  __bf16* Rb    = (__bf16*)take((size_t)M_R * FIN * 2);
  __bf16* W1t   = (__bf16*)take((size_t)FIN * FOUT * 2);   // FOUT x FIN
  __bf16* W2t   = (__bf16*)take((size_t)FIN * FOUT * 2);   // FOUT x FIN
  float*  h1    = (float*)take((size_t)M_R * FOUT * 4);
  float*  h2    = (float*)take((size_t)N_S * FOUT * 4);
  __bf16* h1bT  = (__bf16*)take((size_t)M_R * FOUT * 2);   // FOUT x M_R
  __bf16* h2bT  = (__bf16*)take((size_t)N_S * FOUT * 2);   // FOUT x N_S
  float*  s1    = (float*)take((size_t)M_R * 4);
  float*  s2    = (float*)take((size_t)N_S * 4);
  float*  rm    = (float*)take((size_t)N_S * 4);
  float*  rl    = (float*)take((size_t)N_S * 4);
  float*  up    = (float*)take((size_t)N_S * FOUT * 4);
  float*  vp    = (float*)take((size_t)M_R * FOUT * 4);
  __bf16* uob   = (__bf16*)take((size_t)N_S * FOUT * 2);
  __bf16* vob   = (__bf16*)take((size_t)M_R * FOUT * 2);
  float*  mu1   = (float*)take(FOUT * 4);
  float*  var1  = (float*)take(FOUT * 4);
  float*  mu2   = (float*)take(FOUT * 4);
  float*  var2  = (float*)take(FOUT * 4);

  // 1. bf16 casts (weights transposed to column-major)
  k_cvt_bf16<<<1024, 256, 0, stream>>>(Sinput, Sb, N_S * FIN);
  k_cvt_bf16<<<1024, 256, 0, stream>>>(Rinput, Rb, M_R * FIN);
  k_cvt_wT<<<512, 256, 0, stream>>>(W1, W1t);
  k_cvt_wT<<<512, 256, 0, stream>>>(W2, W2t);

  // 2. input projections (WMMA bf16, f32 accumulate); bf16 copies transposed
  k_gemm_xw<<<dim3(M_R / 64, 4), 128, 0, stream>>>(Rb, W1t, h1, h1bT, M_R);
  k_gemm_xw<<<dim3(N_S / 64, 4), 128, 0, stream>>>(Sb, W2t, h2, h2bT, N_S);

  // 3. attention logits s1 = h1@a1, s2 = h2@a2
  k_rowdot<<<M_R / 4, 128, 0, stream>>>(h1, avec, s1, M_R);
  k_rowdot<<<N_S / 4, 128, 0, stream>>>(h2, avec + FOUT, s2, N_S);

  // 4. per-row masked-softmax stats (max, sum)
  k_softmax_stats<<<N_S, 256, 0, stream>>>(adj, s1, s2, rm, rl);

  // 5. attention GEMMs with on-the-fly att tiles (never materialize 128 MB att)
  k_att_h1<<<N_S / 16, 128, 0, stream>>>(adj, s1, s2, rm, rl, h1bT, up);
  k_attT_h2<<<M_R / 16, 128, 0, stream>>>(adj, s1, s2, rm, rl, h2bT, vp);

  // 6. batch-norm (train stats) + leaky, to bf16
  k_bn_stats<<<FOUT, 256, 0, stream>>>(vp, M_R, mu1, var1);
  k_bn_stats<<<FOUT, 256, 0, stream>>>(up, N_S, mu2, var2);
  k_bn_apply<<<2048, 256, 0, stream>>>(vp, M_R * FOUT, mu1, var1, bn1_g, bn1_b, vob);
  k_bn_apply<<<2048, 256, 0, stream>>>(up, N_S * FOUT, mu2, var2, bn2_g, bn2_b, uob);

  // 7. h_prime = u_out @ v_out.T, then ELU
  k_uvT_elu<<<dim3(N_S / 16, M_R / 256), 128, 0, stream>>>(uob, vob, out);
}
